// StandardAttention_82033875354271
// MI455X (gfx1250) — matrix operand order repaired and checked by reading, not compiled, not run on hardware
//
#include <hip/hip_runtime.h>
#include <stdint.h>

// Problem constants (match reference): B=2, S=2048, E=2048, H=16, D=128
#define B_NUM 2
#define S_DIM 2048
#define E_DIM 2048
#define H_NUM 16
#define D_DIM 128
#define BS    (B_NUM * S_DIM)   // 4096 rows

typedef __attribute__((ext_vector_type(16))) __bf16 v16bf;
typedef __attribute__((ext_vector_type(8)))  float  v8f;

union FragBF { uint4 u[2]; v16bf v; };

static __device__ __forceinline__ unsigned short f32_to_bf16_rne(float f) {
  unsigned int u = __float_as_uint(f);
  u += 0x7FFFu + ((u >> 16) & 1u);          // round-to-nearest-even
  return (unsigned short)(u >> 16);
}

static __device__ __forceinline__ v8f v8f_zero() {
  v8f z = {0.f, 0.f, 0.f, 0.f, 0.f, 0.f, 0.f, 0.f};
  return z;
}

// Generic shared-pointer -> LDS byte address (ISA 10.2: LDS aperture keeps the
// workgroup-relative LDS offset in ADDR[31:0]).
static __device__ __forceinline__ unsigned lds_addr32(const void* p) {
  return (unsigned)(uintptr_t)p;
}

// CDNA5 async copy: global -> LDS, 16B per lane, tracked by ASYNCcnt.
static __device__ __forceinline__ void async_copy_b128(unsigned lds_off,
                                                       const void* gptr) {
  asm volatile("global_load_async_to_lds_b128 %0, %1, off"
               :: "v"(lds_off), "v"((unsigned long long)(uintptr_t)gptr)
               : "memory");
}

static __device__ __forceinline__ void wait_async0() {
  asm volatile("s_wait_asynccnt 0x0" ::: "memory");
}

// ---------------------------------------------------------------------------
// Stage 1a: fp32 -> bf16 elementwise conversion
// ---------------------------------------------------------------------------
__global__ __launch_bounds__(256) void cvt_bf16_kernel(
    const float* __restrict__ src, unsigned short* __restrict__ dst, int n) {
  int i = blockIdx.x * 256 + threadIdx.x;
  if (i < n) dst[i] = f32_to_bf16_rne(src[i]);
}

// ---------------------------------------------------------------------------
// Stage 1b: fp32 -> bf16 transpose (Wt[k*E+n] = W[n*E+k]) via LDS tile
// ---------------------------------------------------------------------------
__global__ __launch_bounds__(256) void transpose_cvt_kernel(
    const float* __restrict__ src, unsigned short* __restrict__ dst) {
  __shared__ float tile[32][33];
  const int bx = blockIdx.x * 32, by = blockIdx.y * 32;
  const int tx = threadIdx.x, ty = threadIdx.y;   // (32, 8)
  #pragma unroll
  for (int i = 0; i < 32; i += 8)
    tile[ty + i][tx] = src[(size_t)(by + ty + i) * E_DIM + bx + tx];
  __syncthreads();
  #pragma unroll
  for (int i = 0; i < 32; i += 8)
    dst[(size_t)(bx + ty + i) * E_DIM + by + tx] = f32_to_bf16_rne(tile[tx][ty + i]);
}

// ---------------------------------------------------------------------------
// Stage 2/4: WMMA GEMM  Y = X @ W^T + bias.
// Workgroup = 128 rows x 128 cols; 8 waves each own 16 rows x 128 cols.
// B chunks (32k x 128n bf16 = 8 KB) are double-buffered in LDS and filled
// with ASYNCcnt-tracked global_load_async_to_lds_b128, shared by all 8 waves.
// MODE 0: Q  -> RoPE, store bf16 [B,H,S,D]
// MODE 1: K  -> RoPE, store bf16 [B,H,D,S] (transposed for QK^T B-fragments)
// MODE 2: V  ->        store bf16 [B,H,S,D]
// MODE 3: O  ->        store fp32 [B,S,E] to d_out
// ---------------------------------------------------------------------------
template <int MODE>
__global__ __launch_bounds__(256) void proj_wmma_kernel(
    const unsigned short* __restrict__ A,    // [BS, E] bf16 activations
    const unsigned short* __restrict__ Bt,   // [E, E] bf16, Bt[k*E+n] = W[n*E+k]
    const float* __restrict__ bias,          // [E]
    const float* __restrict__ cosT,          // [S, D]
    const float* __restrict__ sinT,          // [S, D]
    unsigned short* __restrict__ out_bf,
    float* __restrict__ out_f32) {
  __shared__ unsigned short ldsB[2][32][128];     // 2 x 8 KB double buffer
  const int tid   = threadIdx.x;
  const int lane  = tid & 31;
  const int wave  = tid >> 5;
  const int row_block = blockIdx.x >> 4;     // [0, 32): 128-row block
  const int strip     = blockIdx.x & 15;     // [0, 16): 128-col strip (== head)
  const int rows0 = row_block * 128 + wave * 16;
  const int n0    = strip * 128;
  const int nl    = lane & 15;
  const int hi8   = (lane >> 4) * 8;

  const unsigned lds_base = lds_addr32(&ldsB[0][0][0]);
  // stage one 32x128 bf16 chunk of Bt into ldsB[buf]; 2 x b128 per thread
  auto stage = [&](int buf, int k0) {
    #pragma unroll
    for (int i = 0; i < 2; ++i) {
      const int s = tid + i * 256;                 // 512 x 16B segments
      const unsigned short* g =
          Bt + (size_t)(k0 + (s >> 4)) * E_DIM + n0 + (s & 15) * 8;
      async_copy_b128(lds_base + (unsigned)(buf * 8192 + s * 16), g);
    }
  };

  v8f acc[8];
  #pragma unroll
  for (int t = 0; t < 8; ++t) acc[t] = v8f_zero();

  const unsigned short* arow = A + (size_t)(rows0 + nl) * E_DIM + hi8;

  stage(0, 0);
  for (int it = 0; it < E_DIM / 32; ++it) {
    const int buf = it & 1;
    wait_async0();                 // my slice of ldsB[buf] has landed
    __syncthreads();               // everyone's slice visible; buf^1 free
    if (it + 1 < E_DIM / 32) stage(buf ^ 1, (it + 1) * 32);  // overlap w/ WMMA

    const int k0 = it * 32;
    FragBF fa;
    fa.u[0] = *reinterpret_cast<const uint4*>(arow + k0);        // k0+hi8+0..7
    fa.u[1] = *reinterpret_cast<const uint4*>(arow + k0 + 16);   // k0+hi8+16..23
    #pragma unroll
    for (int t = 0; t < 8; ++t) {
      FragBF fb;                                  // lane%16 = N column of the chunk
      #pragma unroll
      for (int e = 0; e < 16; ++e)
        fb.v[e] = __builtin_bit_cast(__bf16, ldsB[buf][hi8 + (e & 7) + ((e >> 3) << 4)][16 * t + nl]);
      acc[t] = __builtin_amdgcn_wmma_f32_16x16x32_bf16(
          false, fa.v, false, fb.v, (short)0, acc[t], false, false);
    }
  }

  // bias
  #pragma unroll
  for (int t = 0; t < 8; ++t) {
    float bv_ = bias[n0 + 16 * t + nl];
    #pragma unroll
    for (int r = 0; r < 8; ++r) acc[t][r] += bv_;
  }

  // fused RoPE (rotate-half): partner of tile t is tile t^4 (d +/- 64)
  if (MODE == 0 || MODE == 1) {
    v8f res[8];
    #pragma unroll
    for (int t = 0; t < 8; ++t) {
      const int d = 16 * t + nl;
      #pragma unroll
      for (int r = 0; r < 8; ++r) {
        const int row = rows0 + r + hi8;
        const int s   = row & (S_DIM - 1);
        const float c  = cosT[(size_t)s * D_DIM + d];
        const float sn = sinT[(size_t)s * D_DIM + d];
        const float partner = (t < 4) ? -acc[t + 4][r] : acc[t - 4][r];
        res[t][r] = acc[t][r] * c + partner * sn;
      }
    }
    #pragma unroll
    for (int t = 0; t < 8; ++t) acc[t] = res[t];
  }

  #pragma unroll
  for (int t = 0; t < 8; ++t) {
    #pragma unroll
    for (int r = 0; r < 8; ++r) {
      const int row = rows0 + r + hi8;       // [0, BS)
      const int b   = row >> 11;             // / S
      const int s   = row & (S_DIM - 1);
      if (MODE == 3) {
        out_f32[(size_t)row * E_DIM + n0 + 16 * t + nl] = acc[t][r];
      } else {
        const int h = strip;
        const int d = 16 * t + nl;
        size_t idx;
        if (MODE == 1) idx = ((size_t)(b * H_NUM + h) * D_DIM + d) * S_DIM + s;
        else           idx = ((size_t)(b * H_NUM + h) * S_DIM + s) * D_DIM + d;
        out_bf[idx] = f32_to_bf16_rne(acc[t][r]);
      }
    }
  }
}

// ---------------------------------------------------------------------------
// Stage 3: causal flash attention.
// A block's 8 waves handle 8 consecutive 16-query tiles of the SAME (b,h),
// so each 32-key K/V chunk is staged once in LDS (async, double-buffered)
// and consumed by all 8 waves.
// ---------------------------------------------------------------------------
__global__ __launch_bounds__(256) void attn_wmma_kernel(
    const unsigned short* __restrict__ q,   // [B,H,S,D] bf16 (RoPE'd)
    const unsigned short* __restrict__ kT,  // [B,H,D,S] bf16 (RoPE'd)
    const unsigned short* __restrict__ v,   // [B,H,S,D] bf16
    unsigned short* __restrict__ o_bf) {    // [B,S,E]  bf16
  __shared__ unsigned short ldsK[2][128][32];   // K^T chunk: [d][key], 2x8 KB
  __shared__ unsigned short ldsV[2][32][128];   // V chunk:   [key][d], 2x8 KB
  __shared__ unsigned short p_lds[8][16][32];   // per-wave P tile (16q x 32k)
  const int tid  = threadIdx.x;
  const int lane = tid & 31;
  const int wave = tid >> 5;
  const int bh   = blockIdx.x >> 4;             // [0, 32)
  const int qt   = (blockIdx.x & 15) * 8 + wave;
  const int q0   = qt * 16;
  const int nl   = lane & 15;
  const int hi8  = (lane >> 4) * 8;
  const float scale = 0.0883883476483184f;      // 1/sqrt(128)

  const unsigned short* kbase = kT + (size_t)bh * D_DIM * S_DIM;
  const unsigned short* vbase = v  + (size_t)bh * S_DIM * D_DIM;

  const unsigned ldsK_base = lds_addr32(&ldsK[0][0][0]);
  const unsigned ldsV_base = lds_addr32(&ldsV[0][0][0]);
  // stage K^T (128d x 32keys) and V (32keys x 128d) chunks for keys [n0,n0+32)
  auto stage_kv = [&](int buf, int n0) {
    #pragma unroll
    for (int i = 0; i < 2; ++i) {
      const int s = tid + i * 256;              // 512 x 16B segments each
      const unsigned short* gk =
          kbase + (size_t)(s >> 2) * S_DIM + n0 + (s & 3) * 8;
      async_copy_b128(ldsK_base + (unsigned)(buf * 8192 + s * 16), gk);
      const unsigned short* gv =
          vbase + (size_t)(n0 + (s >> 4)) * D_DIM + (s & 15) * 8;
      async_copy_b128(ldsV_base + (unsigned)(buf * 8192 + s * 16), gv);
    }
  };

  // preload Q fragments for the whole head dim (4 K-chunks of 32)
  FragBF aq[4];
  const unsigned short* qrow = q + ((size_t)bh * S_DIM + q0 + nl) * D_DIM + hi8;
  #pragma unroll
  for (int c = 0; c < 4; ++c) {
    aq[c].u[0] = *reinterpret_cast<const uint4*>(qrow + 32 * c);
    aq[c].u[1] = *reinterpret_cast<const uint4*>(qrow + 32 * c + 16);
  }

  v8f o[8];
  #pragma unroll
  for (int t = 0; t < 8; ++t) o[t] = v8f_zero();
  float rm[8], rl[8];
  #pragma unroll
  for (int r = 0; r < 8; ++r) { rm[r] = -1e30f; rl[r] = 0.f; }

  const int my_nb = (q0 + 47) >> 5;                           // causal blocks
  const int q0_last = ((blockIdx.x & 15) * 8 + 7) * 16;
  const int nb_max  = (q0_last + 47) >> 5;                    // uniform bound

  stage_kv(0, 0);
  for (int kb = 0; kb < nb_max; ++kb) {
    const int buf = kb & 1;
    wait_async0();
    __syncthreads();
    if (kb + 1 < nb_max) stage_kv(buf ^ 1, (kb + 1) * 32);

    if (kb < my_nb) {
      const int n0 = kb * 32;
      // S = Q @ K^T for keys [n0, n0+32): two 16x16 score tiles
      v8f s0 = v8f_zero(), s1 = v8f_zero();
      #pragma unroll
      for (int c = 0; c < 4; ++c) {
        FragBF b0, b1;
        #pragma unroll
        for (int e = 0; e < 16; ++e) {
          const int kd = c * 32 + hi8 + (e & 7) + ((e >> 3) << 4);
          b0.v[e] = __builtin_bit_cast(__bf16, ldsK[buf][kd][nl]);
          b1.v[e] = __builtin_bit_cast(__bf16, ldsK[buf][kd][nl + 16]);
        }
        s0 = __builtin_amdgcn_wmma_f32_16x16x32_bf16(false, aq[c].v, false, b0.v, (short)0, s0, false, false);
        s1 = __builtin_amdgcn_wmma_f32_16x16x32_bf16(false, aq[c].v, false, b1.v, (short)0, s1, false, false);
      }

      // online softmax per row (rows live across 16-lane halves of each VGPR)
      float corr[8];
      #pragma unroll
      for (int r = 0; r < 8; ++r) {
        const int qrow_i = q0 + r + hi8;
        const int key0   = n0 + nl;
        float x0 = s0[r] * scale;
        float x1 = s1[r] * scale;
        if (key0 > qrow_i)      x0 = -1e9f;       // causal mask
        if (key0 + 16 > qrow_i) x1 = -1e9f;
        float mx = fmaxf(x0, x1);
        #pragma unroll
        for (int off = 1; off < 16; off <<= 1) mx = fmaxf(mx, __shfl_xor(mx, off, 32));
        const float mnew = fmaxf(rm[r], mx);
        const float c  = __expf(rm[r] - mnew);
        const float p0 = __expf(x0 - mnew);
        const float p1 = __expf(x1 - mnew);
        float ps = p0 + p1;
        #pragma unroll
        for (int off = 1; off < 16; off <<= 1) ps += __shfl_xor(ps, off, 32);
        rl[r] = rl[r] * c + ps;
        rm[r] = mnew;
        corr[r] = c;
        // stage P tile (C-layout -> LDS) for re-fragmentation into A-layout
        p_lds[wave][r + hi8][nl]      = f32_to_bf16_rne(p0);
        p_lds[wave][r + hi8][nl + 16] = f32_to_bf16_rne(p1);
      }
      #pragma unroll
      for (int t = 0; t < 8; ++t) {
        #pragma unroll
        for (int r = 0; r < 8; ++r) o[t][r] *= corr[r];
      }

      // P as A-fragment from LDS (per-wave region; DScnt keeps order in-wave)
      FragBF pa;
      pa.u[0] = *reinterpret_cast<const uint4*>(&p_lds[wave][nl][hi8]);
      pa.u[1] = *reinterpret_cast<const uint4*>(&p_lds[wave][nl][hi8 + 16]);

      // O += P @ V
      #pragma unroll
      for (int t = 0; t < 8; ++t) {
        FragBF fb;                                // lane%16 = d column of chunk
        #pragma unroll
        for (int e = 0; e < 16; ++e)
          fb.v[e] = __builtin_bit_cast(__bf16, ldsV[buf][hi8 + (e & 7) + ((e >> 3) << 4)][16 * t + nl]);
        o[t] = __builtin_amdgcn_wmma_f32_16x16x32_bf16(false, pa.v, false, fb.v, (short)0, o[t], false, false);
      }
    }
  }

  // normalize + store bf16 [B, S, E] for the output projection
  const int b = bh >> 4, h = bh & 15;
  float rinv[8];
  #pragma unroll
  for (int r = 0; r < 8; ++r) rinv[r] = 1.0f / rl[r];
  #pragma unroll
  for (int t = 0; t < 8; ++t) {
    #pragma unroll
    for (int r = 0; r < 8; ++r) {
      const int s = q0 + r + hi8;
      o_bf[(size_t)(b * S_DIM + s) * E_DIM + h * D_DIM + 16 * t + nl] =
          f32_to_bf16_rne(o[t][r] * rinv[r]);
    }
  }
}

// ---------------------------------------------------------------------------
// Launcher. Workspace budget (bf16 staging): 2*16MB x + 4*8MB wT + 3*16MB qkv
// + 16MB attn = 128 MB.
// ---------------------------------------------------------------------------
extern "C" void kernel_launch(void* const* d_in, const int* in_sizes, int n_in,
                              void* d_out, int out_size, void* d_ws, size_t ws_size,
                              hipStream_t stream) {
  const float* x_q  = (const float*)d_in[0];
  const float* x_kv = (const float*)d_in[1];
  const float* cosT = (const float*)d_in[2];
  const float* sinT = (const float*)d_in[3];
  const float* wq = (const float*)d_in[4];
  const float* bq = (const float*)d_in[5];
  const float* wk = (const float*)d_in[6];
  const float* bk = (const float*)d_in[7];
  const float* wv = (const float*)d_in[8];
  const float* bv = (const float*)d_in[9];
  const float* wo = (const float*)d_in[10];
  const float* bo = (const float*)d_in[11];
  float* out = (float*)d_out;

  unsigned char* ws = (unsigned char*)d_ws;
  size_t off = 0;
  auto alloc = [&](size_t bytes) -> unsigned short* {
    unsigned short* p = (unsigned short*)(ws + off);
    off += (bytes + 255) & ~(size_t)255;
    return p;
  };
  const size_t actB = (size_t)BS * E_DIM * 2;     // 16 MB
  const size_t wB   = (size_t)E_DIM * E_DIM * 2;  // 8 MB
  unsigned short* xq_bf  = alloc(actB);
  unsigned short* xkv_bf = alloc(actB);
  unsigned short* wqT = alloc(wB);
  unsigned short* wkT = alloc(wB);
  unsigned short* wvT = alloc(wB);
  unsigned short* woT = alloc(wB);
  unsigned short* qb   = alloc(actB);   // [B,H,S,D]
  unsigned short* kTb  = alloc(actB);   // [B,H,D,S]
  unsigned short* vb   = alloc(actB);   // [B,H,S,D]
  unsigned short* attn = alloc(actB);   // [B,S,E]

  const int nAct = BS * E_DIM;                    // 8388608, multiple of 256
  cvt_bf16_kernel<<<nAct / 256, 256, 0, stream>>>(x_q,  xq_bf,  nAct);
  cvt_bf16_kernel<<<nAct / 256, 256, 0, stream>>>(x_kv, xkv_bf, nAct);

  dim3 tb(32, 8), tg(E_DIM / 32, E_DIM / 32);
  transpose_cvt_kernel<<<tg, tb, 0, stream>>>(wq, wqT);
  transpose_cvt_kernel<<<tg, tb, 0, stream>>>(wk, wkT);
  transpose_cvt_kernel<<<tg, tb, 0, stream>>>(wv, wvT);
  transpose_cvt_kernel<<<tg, tb, 0, stream>>>(wo, woT);

  // 512 workgroups: 32 row-blocks x 16 col-strips, 8 waves each
  proj_wmma_kernel<0><<<512, 256, 0, stream>>>(xq_bf,  wqT, bq, cosT, sinT, qb,  nullptr);
  proj_wmma_kernel<1><<<512, 256, 0, stream>>>(xkv_bf, wkT, bk, cosT, sinT, kTb, nullptr);
  proj_wmma_kernel<2><<<512, 256, 0, stream>>>(xkv_bf, wvT, bv, cosT, sinT, vb,  nullptr);

  // 512 workgroups: 32 (b,h) x 16 q-tile groups of 8 waves
  attn_wmma_kernel<<<512, 256, 0, stream>>>(qb, kTb, vb, attn);

  proj_wmma_kernel<3><<<512, 256, 0, stream>>>(attn, woT, bo, cosT, sinT, nullptr, out);
}
